// MoEBlock_42399917146873
// MI455X (gfx1250) — matrix operand order repaired and checked by reading, not compiled, not run on hardware
//
#include <hip/hip_runtime.h>
#include <cstdint>
#include <cstddef>

// ---------------------------------------------------------------------------
// Types for CDNA5 WMMA (wave32): v_wmma_f32_16x16x32_bf16
// ---------------------------------------------------------------------------
typedef __attribute__((ext_vector_type(16))) __bf16 v16bf;
typedef __attribute__((ext_vector_type(8)))  __bf16 v8bf;
typedef __attribute__((ext_vector_type(8)))  float  v8f;

union FragAB {
    v16bf v;
    v8bf  h[2];
};

__device__ __forceinline__ __bf16 f2bf(float f) {
    union { float f; uint32_t u; } c; c.f = f;
    uint32_t u = c.u;
    uint32_t r = u + 0x7FFFu + ((u >> 16) & 1u);   // round-to-nearest-even
    union { uint16_t s; __bf16 b; } o; o.s = (uint16_t)(r >> 16);
    return o.b;
}

#define ALPHA 1.702f
#define LIMIT 7.0f
#define EPS   1e-6f

// ---------------------------------------------------------------------------
// Kernel 1: RMSNorm + router (one 256-thread block per token)
// ---------------------------------------------------------------------------
__global__ __launch_bounds__(256) void rmsnorm_router_kernel(
    const float* __restrict__ x, const float* __restrict__ scale,
    const float* __restrict__ gate_w,
    __bf16* __restrict__ xn, float* __restrict__ probs, int* __restrict__ eids,
    int H, int E)
{
    __shared__ float red[256];
    __shared__ float xs[2048];     // normalized row (H <= 2048)
    __shared__ float logits[32];

    const int tid = threadIdx.x;
    const long long t = blockIdx.x;
    const float* xr = x + t * (long long)H;

    float ss = 0.f;
    for (int i = tid; i < H; i += 256) { float v = xr[i]; ss += v * v; }
    red[tid] = ss; __syncthreads();
    for (int off = 128; off > 0; off >>= 1) {
        if (tid < off) red[tid] += red[tid + off];
        __syncthreads();
    }
    float inv = rsqrtf(red[0] / (float)H + EPS);
    __syncthreads();

    for (int i = tid; i < H; i += 256) {
        float n = xr[i] * inv * scale[i];
        xs[i] = n;
        xn[t * (long long)H + i] = f2bf(n);
    }
    __syncthreads();

    for (int e = 0; e < E; ++e) {
        float p = 0.f;
        const float* gw = gate_w + (long long)e * H;
        for (int i = tid; i < H; i += 256) p += xs[i] * gw[i];
        red[tid] = p; __syncthreads();
        for (int off = 128; off > 0; off >>= 1) {
            if (tid < off) red[tid] += red[tid + off];
            __syncthreads();
        }
        if (tid == 0) logits[e] = red[0];
        __syncthreads();
    }

    if (tid == 0) {
        float lv[32]; bool taken[32];
        for (int e = 0; e < E; ++e) { lv[e] = logits[e]; taken[e] = false; }
        float val[4]; int idx[4];
        for (int i = 0; i < 4; ++i) {
            int bi = 0; float bv = -3.4e38f;
            for (int e = 0; e < E; ++e)
                if (!taken[e] && lv[e] > bv) { bv = lv[e]; bi = e; }
            taken[bi] = true; val[i] = bv; idx[i] = bi;
        }
        float m = val[0], s = 0.f, pv[4];
        for (int i = 0; i < 4; ++i) { pv[i] = expf(val[i] - m); s += pv[i]; }
        for (int i = 0; i < 4; ++i) probs[t * 4 + i] = pv[i] / s;
        if (t == 0)
            for (int i = 0; i < 4; ++i) eids[i] = idx[i] % E;
    }
}

// ---------------------------------------------------------------------------
// Kernel 2: out = x  (residual init)
// ---------------------------------------------------------------------------
__global__ void init_out_kernel(const float* __restrict__ x,
                                float* __restrict__ out, long long n) {
    long long i = (long long)blockIdx.x * blockDim.x + threadIdx.x;
    long long stride = (long long)gridDim.x * blockDim.x;
    for (; i < n; i += stride) out[i] = x[i];
}

// ---------------------------------------------------------------------------
// Kernel 3: convert one expert's fp32 weights -> bf16 workspace
// ---------------------------------------------------------------------------
__global__ void cvt_expert_kernel(const float* __restrict__ src,
                                  const int* __restrict__ eids, int which,
                                  long long per, __bf16* __restrict__ dst) {
    const float* s = src + (long long)eids[which] * per;
    long long i = (long long)blockIdx.x * blockDim.x + threadIdx.x;
    long long stride = (long long)gridDim.x * blockDim.x;
    for (; i < per; i += stride) dst[i] = f2bf(s[i]);
}

// ---------------------------------------------------------------------------
// Kernel 4: up-projection + fused SwiGLU.
//   act[M, I] = swiglu( xn[M,H] @ wu[2I,H]^T + bu )
//   Block tile: 128 rows x 128 act cols (g AND l halves computed).
//   8 waves (2x4), each wave 64x32: 16 WMMA per 16 b128 loads per k-step.
// ---------------------------------------------------------------------------
__global__ __launch_bounds__(256) void up_swiglu_kernel(
    const __bf16* __restrict__ A,   // xn [M, H]
    const __bf16* __restrict__ W,   // wu_bf16 [2I, H]
    const float* __restrict__ b_up, // [E, 2I]
    const int* __restrict__ eids, int which,
    __bf16* __restrict__ act,       // [M, I]
    int H, int I)
{
    const int eid = eids[which];
    const float* bu = b_up + (long long)eid * (2 * I);

    const int wave = threadIdx.x >> 5;
    const int lane = threadIdx.x & 31;
    const int wm = wave >> 2;            // 0..1
    const int wn = wave & 3;             // 0..3
    const int lh = (lane >> 4) & 1;      // lane half
    const int lr = lane & 15;

    const int m0 = blockIdx.x * 128 + wm * 64;
    const int nb = blockIdx.y * 128 + wn * 32;   // act col base for this wave

    v8f cg[4][2], cl[4][2];
#pragma unroll
    for (int s = 0; s < 4; ++s)
#pragma unroll
        for (int j = 0; j < 2; ++j) { cg[s][j] = (v8f)0.f; cl[s][j] = (v8f)0.f; }

    const long long ldw = H;
    for (int k0 = 0; k0 < H; k0 += 32) {
        FragAB bg[2], bl[2], a[4];
#pragma unroll
        for (int j = 0; j < 2; ++j) {
            const __bf16* pg = W + (long long)(nb + j * 16 + lr) * ldw + k0 + lh * 16;
            bg[j].h[0] = *(const v8bf*)pg;
            bg[j].h[1] = *(const v8bf*)(pg + 8);
            const __bf16* pl = W + (long long)(nb + I + j * 16 + lr) * ldw + k0 + lh * 16;
            bl[j].h[0] = *(const v8bf*)pl;
            bl[j].h[1] = *(const v8bf*)(pl + 8);
        }
#pragma unroll
        for (int s = 0; s < 4; ++s) {
            const __bf16* pa = A + (long long)(m0 + s * 16 + lr) * ldw + k0 + lh * 16;
            a[s].h[0] = *(const v8bf*)pa;
            a[s].h[1] = *(const v8bf*)(pa + 8);
        }
#pragma unroll
        for (int s = 0; s < 4; ++s) {
#pragma unroll
            for (int j = 0; j < 2; ++j) {
                cg[s][j] = __builtin_amdgcn_wmma_f32_16x16x32_bf16(
                    false, a[s].v, false, bg[j].v, (short)0, cg[s][j], false, false);
                cl[s][j] = __builtin_amdgcn_wmma_f32_16x16x32_bf16(
                    false, a[s].v, false, bl[j].v, (short)0, cl[s][j], false, false);
            }
        }
    }

    // epilogue: SwiGLU, write bf16 act
#pragma unroll
    for (int j = 0; j < 2; ++j) {
        const int col = nb + j * 16 + lr;
        const float bug = bu[col];
        const float bul = bu[col + I];
#pragma unroll
        for (int s = 0; s < 4; ++s) {
#pragma unroll
            for (int r = 0; r < 8; ++r) {
                const int row = m0 + s * 16 + r + lh * 8;
                float g = cg[s][j][r] + bug;
                float l = cl[s][j][r] + bul;
                g = fminf(g, LIMIT);
                l = fminf(fmaxf(l, -LIMIT), LIMIT);
                float sig = 1.0f / (1.0f + expf(-ALPHA * g));
                act[(long long)row * I + col] = f2bf(g * sig * (l + 1.0f));
            }
        }
    }
}

// ---------------------------------------------------------------------------
// Kernel 5: down-projection, scaled accumulate into out.
//   out[M,H] += probs[:,which] * ( act[M,I] @ wd[H,I]^T + bd )
//   Block tile: 128 x 128, 8 waves (2x4), each wave 64x32.
// ---------------------------------------------------------------------------
__global__ __launch_bounds__(256) void down_acc_kernel(
    const __bf16* __restrict__ A,     // act [M, I]
    const __bf16* __restrict__ W,     // wd_bf16 [H, I]
    const float* __restrict__ b_down, // [E, H]
    const int* __restrict__ eids, int which,
    const float* __restrict__ probs,  // [M, 4]
    float* __restrict__ out,          // [M, H]
    int H, int I)
{
    const int eid = eids[which];
    const float* bd = b_down + (long long)eid * H;

    const int wave = threadIdx.x >> 5;
    const int lane = threadIdx.x & 31;
    const int wm = wave >> 2;
    const int wn = wave & 3;
    const int lh = (lane >> 4) & 1;
    const int lr = lane & 15;

    const int m0 = blockIdx.x * 128 + wm * 64;
    const int nb = blockIdx.y * 128 + wn * 32;

    v8f c[4][2];
#pragma unroll
    for (int s = 0; s < 4; ++s)
#pragma unroll
        for (int j = 0; j < 2; ++j) c[s][j] = (v8f)0.f;

    const long long lda = I;
    for (int k0 = 0; k0 < I; k0 += 32) {
        FragAB b[2], a[4];
#pragma unroll
        for (int j = 0; j < 2; ++j) {
            const __bf16* pw = W + (long long)(nb + j * 16 + lr) * lda + k0 + lh * 16;
            b[j].h[0] = *(const v8bf*)pw;
            b[j].h[1] = *(const v8bf*)(pw + 8);
        }
#pragma unroll
        for (int s = 0; s < 4; ++s) {
            const __bf16* pa = A + (long long)(m0 + s * 16 + lr) * lda + k0 + lh * 16;
            a[s].h[0] = *(const v8bf*)pa;
            a[s].h[1] = *(const v8bf*)(pa + 8);
        }
#pragma unroll
        for (int s = 0; s < 4; ++s)
#pragma unroll
            for (int j = 0; j < 2; ++j)
                c[s][j] = __builtin_amdgcn_wmma_f32_16x16x32_bf16(
                    false, a[s].v, false, b[j].v, (short)0, c[s][j], false, false);
    }

    const int col0 = nb + lr;
    const float bdv0 = bd[col0];
    const float bdv1 = bd[col0 + 16];
#pragma unroll
    for (int s = 0; s < 4; ++s) {
#pragma unroll
        for (int r = 0; r < 8; ++r) {
            const int row = m0 + s * 16 + r + lh * 8;
            const float p = probs[(long long)row * 4 + which];
            const long long base = (long long)row * H;
            out[base + col0]      += p * (c[s][0][r] + bdv0);
            out[base + col0 + 16] += p * (c[s][1][r] + bdv1);
        }
    }
}

// ---------------------------------------------------------------------------
// Host launcher
// ---------------------------------------------------------------------------
extern "C" void kernel_launch(void* const* d_in, const int* in_sizes, int n_in,
                              void* d_out, int out_size, void* d_ws, size_t ws_size,
                              hipStream_t stream) {
    const float* x     = (const float*)d_in[0];
    const float* scale = (const float*)d_in[1];
    const float* gw    = (const float*)d_in[2];
    const float* wu    = (const float*)d_in[3];
    const float* bu    = (const float*)d_in[4];
    const float* wd    = (const float*)d_in[5];
    const float* bd    = (const float*)d_in[6];
    float* out = (float*)d_out;

    const int H        = in_sizes[1];
    const int E        = in_sizes[2] / H;
    const long long I2 = (long long)in_sizes[4] / E;
    const long long I  = I2 / 2;
    const long long M  = (long long)in_sizes[0] / H;

    char* ws = (char*)d_ws;
    size_t off = 0;
    auto alloc = [&](size_t bytes) -> void* {
        off = (off + 255) & ~(size_t)255;
        void* p = ws + off;
        off += bytes;
        return p;
    };
    __bf16* xn    = (__bf16*)alloc((size_t)M * H * 2);
    __bf16* act   = (__bf16*)alloc((size_t)M * I * 2);
    __bf16* wub   = (__bf16*)alloc((size_t)I2 * H * 2);
    __bf16* wdb   = (__bf16*)alloc((size_t)H * I * 2);
    float*  probs = (float*)alloc((size_t)M * 4 * sizeof(float));
    int*    eids  = (int*)alloc(64);

    rmsnorm_router_kernel<<<(unsigned)M, 256, 0, stream>>>(
        x, scale, gw, xn, probs, eids, H, E);
    init_out_kernel<<<2048, 256, 0, stream>>>(x, out, M * H);

    for (int i = 0; i < 4; ++i) {
        cvt_expert_kernel<<<4096, 256, 0, stream>>>(wu, eids, i, I2 * H, wub);
        cvt_expert_kernel<<<4096, 256, 0, stream>>>(wd, eids, i, H * I, wdb);
        dim3 gu((unsigned)(M / 128), (unsigned)(I / 128));
        up_swiglu_kernel<<<gu, 256, 0, stream>>>(xn, wub, bu, eids, i, act, H, (int)I);
        dim3 gd((unsigned)(M / 128), (unsigned)(H / 128));
        down_acc_kernel<<<gd, 256, 0, stream>>>(act, wdb, bd, eids, i, probs, out, H, (int)I);
    }
}
